// MultiHeadConv_28724741276040
// MI455X (gfx1250) — compile-verified, matrix operand order checked by reading, not run on hardware
//
#include <hip/hip_runtime.h>

typedef _Float16 f16;
typedef __attribute__((ext_vector_type(16))) _Float16 v16h;
typedef __attribute__((ext_vector_type(8)))  _Float16 v8h;
typedef __attribute__((ext_vector_type(8)))  float    v8f;

#define HEADS 8
#define HD    32
#define HW    4096
#define CCH   256

// per-lane async 16B copy global->LDS; the 24-bit INST_OFFSET is applied to
// BOTH the LDS and global addresses, so congruent tiles need only one base.
#define ASYNC_LD_B128(lds_off_u32, gptr, ioff)                              \
    asm volatile("global_load_async_to_lds_b128 %0, %1, off offset:%c2"     \
                 :: "v"(lds_off_u32),                                       \
                    "v"((unsigned long long)(uintptr_t)(gptr)),             \
                    "i"(ioff) : "memory")
#define WAIT_ASYNCCNT(n) asm volatile("s_wait_asynccnt %0" :: "i"(n) : "memory")

// ---------------------------------------------------------------------------
// Kernel 1: pointwise GEMM  y[cout, s] = pw_w[cout,:] . (x[:,s]*dw_w + dw_b) + pw_b
// Grid: (64 spatial tiles of 64, 4 cout blocks of 64). Block = 128 thr = 4 waves.
// Each wave owns 16 couts x 64 spatial (1 A-frag reused over 4 B-frags -> 4 WMMA
// per K-step). xhat chunk staged transposed in LDS; output repacked through LDS
// for coalesced b128 stores. ADD_POS folds posH+posW into K (q.k+q.pos==q.(k+pos)).
// ---------------------------------------------------------------------------
template<int ADD_POS>
__global__ __launch_bounds__(128)
void qkv_gemm(const float* __restrict__ x,
              const float* __restrict__ dw_w, const float* __restrict__ dw_b,
              const float* __restrict__ pw_w, const float* __restrict__ pw_b,
              const float* __restrict__ posH, const float* __restrict__ posW,
              const int* __restrict__ patch_idx,
              f16* __restrict__ out)
{
    __shared__ __align__(16) f16 lsbuf[64 * 64];  // staging (2K used) / out repack (4K)
    f16* xT = lsbuf;                              // [s_local][cin_local]

    const int st   = blockIdx.x;            // spatial tile (64 cols)
    const int cb   = blockIdx.y;            // cout block (64 rows)
    const int t    = threadIdx.x;
    const int wave = t >> 5;
    const int lane = t & 31;
    const int s0   = st * 64;
    const int c0w  = cb * 64 + wave * 16;   // this wave's cout base
    const int lrow = lane & 15;
    const int koff = (lane >> 4) * 8;       // K sub-block per lane half

    v8f acc[4] = {};

    #pragma unroll 2
    for (int kc = 0; kc < CCH; kc += 32) {
        __syncthreads();
        // cooperative stage of xhat^T: 2048 f16, 16 per thread, coalesced reads
        #pragma unroll
        for (int j = 0; j < 16; ++j) {
            int e    = t + 128 * j;         // 0..2047
            int ci   = e >> 6;              // 0..31
            int sl   = e & 63;              // 0..63
            int cing = kc + ci;
            float xh = x[cing * HW + s0 + sl] * dw_w[cing] + dw_b[cing];
            xT[sl * 32 + ci] = (f16)xh;
        }
        __syncthreads();

        // A fragment: weights row (cout = c0w+lrow), K = koff..+7, koff+16..+23
        const float* wr = pw_w + (size_t)(c0w + lrow) * CCH + kc + koff;
        v16h a;
        #pragma unroll
        for (int i = 0; i < 8; ++i) { a[i] = (f16)wr[i]; a[8 + i] = (f16)wr[16 + i]; }

        // 4 B fragments (columns s = nt*16+lrow are K-contiguous after transpose)
        #pragma unroll
        for (int nt = 0; nt < 4; ++nt) {
            v8h b0 = *(const v8h*)(&xT[(nt * 16 + lrow) * 32 + koff]);
            v8h b1 = *(const v8h*)(&xT[(nt * 16 + lrow) * 32 + koff + 16]);
            v16h b;
            #pragma unroll
            for (int i = 0; i < 8; ++i) { b[i] = b0[i]; b[8 + i] = b1[i]; }
            acc[nt] = __builtin_amdgcn_wmma_f32_16x16x32_f16(false, a, false, b,
                                                             (short)0, acc[nt], false, false);
        }
    }

    // epilogue: bias (+ positional term for K), repack via LDS, b128 stores
    int h_base = 0, w_base = 0;
    if (ADD_POS) {
        const int pidx = patch_idx[0];
        w_base = (pidx % 8) * 64;
        h_base = (pidx / 8) * 64;
    }
    __syncthreads();                         // xT consumers done; reuse lsbuf
    #pragma unroll
    for (int nt = 0; nt < 4; ++nt) {
        const int sg = s0 + nt * 16 + lrow;  // N = lane%16
        #pragma unroll
        for (int r = 0; r < 8; ++r) {
            int m  = r + 8 * (lane >> 4);    // D layout: M = r + 8*(lane/16)
            int cg = c0w + m;
            float y = acc[nt][r] + pw_b[cg];
            if (ADD_POS) {
                y += posH[cg * 512 + h_base + (sg >> 6)]
                   + posW[cg * 512 + w_base + (sg & 63)];
            }
            lsbuf[(wave * 16 + m) * 64 + nt * 16 + lrow] = (f16)y;
        }
    }
    __syncthreads();
    {   // cooperative coalesced store: thread -> 64B of one cout row
        const int row  = t >> 1;             // 0..63
        const int half = t & 1;              // 0 / 1 (32 halves each)
        const f16* src = &lsbuf[row * 64 + half * 32];
        f16* dst = out + (size_t)(cb * 64 + row) * HW + s0 + half * 32;
        #pragma unroll
        for (int j = 0; j < 4; ++j)
            *(v8h*)(dst + j * 8) = *(const v8h*)(src + j * 8);
    }
}

// ---------------------------------------------------------------------------
// Kernel 2: fused attention.  Block = 256 thr = 8 waves = 8 heads; one block
// per 16-row n-tile.  Softmax is across the HEAD axis (reference quirk) ->
// elementwise per (n,m): one LDS exchange + barrier per 64-wide m-chunk.
// V tiles arrive via double-buffered GLOBAL_LOAD_ASYNC_TO_LDS_B128 (single
// base address + immediate offsets, congruent LDS/global layout) overlapped
// with QK WMMA + softmax; scores never touch HBM (512 MB avoided).
// ---------------------------------------------------------------------------
__global__ __launch_bounds__(256)
void fused_attn(const f16* __restrict__ q,
                const f16* __restrict__ k,
                const f16* __restrict__ v,
                float* __restrict__ out)
{
    __shared__ __align__(16) float sexp[HEADS * 1024];         // [head][sc][r*32+lane]
    __shared__ __align__(16) f16   attnb[HEADS * 16 * 64];     // per-wave attn tile
    __shared__ __align__(16) f16   vstg[HEADS * 2 * 64 * 32];  // per-wave V dbl-buf

    const int n0   = blockIdx.x * 16;
    const int h    = threadIdx.x >> 5;      // wave == head
    const int lane = threadIdx.x & 31;
    const int lrow = lane & 15;
    const int koff = (lane >> 4) * 8;
    const size_t hoff = (size_t)h * (HW * HD);

    // persistent Q A-fragment: row n0+lrow is K-contiguous -> two b128 loads
    const f16* qr = q + hoff + (size_t)(n0 + lrow) * HD + koff;
    v16h qa;
    {
        v8h lo = *(const v8h*)qr;
        v8h hi = *(const v8h*)(qr + 16);
        #pragma unroll
        for (int i = 0; i < 8; ++i) { qa[i] = lo[i]; qa[8 + i] = hi[i]; }
    }

    v8f acc0 = {};            // out[n-tile, d=0..15]
    v8f acc1 = {};            // out[n-tile, d=16..31]
    float* myexp  = &sexp[h * 1024];
    f16*   myattn = &attnb[h * 1024];
    f16*   myv0   = &vstg[h * 4096];
    f16*   myv1   = myv0 + 2048;

    // async DMA of one 64x32 f16 V tile: 2 rows/lane, one base, imm offsets
    auto issueV = [&](f16* buf, int mchunk) {
        const f16* vr = v + hoff + (size_t)(mchunk + lane) * HD;
        unsigned lds0 = (unsigned)(uintptr_t)(&buf[lane * 32]);
        #pragma unroll
        for (int rr = 0; rr < 2; ++rr)
            #pragma unroll
            for (int j = 0; j < 4; ++j)
                ASYNC_LD_B128(lds0, vr, rr * 2048 + j * 16);
    };

    // one 64-wide m-chunk: 4x QK WMMA + cross-head softmax + 4x PV WMMA
    auto doChunk = [&](int m0c, const f16* vbuf) {
        float e[4][8];
        #pragma unroll
        for (int sc = 0; sc < 4; ++sc) {
            const int m1 = m0c + sc * 16;
            // K B-fragment: column m = K-row m1+lrow, K-contiguous in memory
            const f16* kr = k + hoff + (size_t)(m1 + lrow) * HD + koff;
            v16h kb;
            {
                v8h lo = *(const v8h*)kr;
                v8h hi = *(const v8h*)(kr + 16);
                #pragma unroll
                for (int i = 0; i < 8; ++i) { kb[i] = lo[i]; kb[8 + i] = hi[i]; }
            }
            v8f zero = {};
            v8f s = __builtin_amdgcn_wmma_f32_16x16x32_f16(false, qa, false, kb,
                                                           (short)0, zero, false, false);
            #pragma unroll
            for (int r = 0; r < 8; ++r) {
                e[sc][r] = __expf(s[r]);
                myexp[sc * 256 + r * 32 + lane] = e[sc][r];
            }
        }
        __syncthreads();
        // softmax across heads: element (sc,r,lane) aligns across all 8 waves
        #pragma unroll
        for (int sc = 0; sc < 4; ++sc) {
            #pragma unroll
            for (int r = 0; r < 8; ++r) {
                const int idx = sc * 256 + r * 32 + lane;
                float den = sexp[idx];
                #pragma unroll
                for (int w2 = 1; w2 < HEADS; ++w2)
                    den += sexp[w2 * 1024 + idx];
                float a = e[sc][r] / den;
                int nl = r + 8 * (lane >> 4);       // C/D layout -> row-major tile
                myattn[nl * 64 + sc * 16 + lrow] = (f16)a;
            }
        }
        __syncthreads();                            // protects sexp reuse + attnb

        // drain this buffer's 8 async loads (in-order completion: <=8 left
        // outstanding means this tile landed; next tile may still be in flight)
        WAIT_ASYNCCNT(8);
        // PV: two K-groups (m 0..31 / 32..63) x two d-halves
        #pragma unroll
        for (int g = 0; g < 2; ++g) {
            v16h aa;
            {
                const f16* ar = &myattn[lrow * 64 + g * 32 + koff];
                v8h lo = *(const v8h*)ar;
                v8h hi = *(const v8h*)(ar + 16);
                #pragma unroll
                for (int i = 0; i < 8; ++i) { aa[i] = lo[i]; aa[8 + i] = hi[i]; }
            }
            #pragma unroll
            for (int dh = 0; dh < 2; ++dh) {
                const int dc = dh * 16 + lrow;
                v16h vb;
                #pragma unroll
                for (int i = 0; i < 8; ++i) {
                    vb[i]     = vbuf[(g * 32 + koff + i) * 32 + dc];
                    vb[8 + i] = vbuf[(g * 32 + koff + 16 + i) * 32 + dc];
                }
                if (dh == 0)
                    acc0 = __builtin_amdgcn_wmma_f32_16x16x32_f16(false, aa, false, vb,
                                                                  (short)0, acc0, false, false);
                else
                    acc1 = __builtin_amdgcn_wmma_f32_16x16x32_f16(false, aa, false, vb,
                                                                  (short)0, acc1, false, false);
            }
        }
    };

    issueV(myv0, 0);                                // prime buffer 0
    for (int m0 = 0; m0 < HW; m0 += 128) {
        __builtin_prefetch(k + hoff + (size_t)(((m0 + 64) & (HW - 1)) + lane) * HD, 0, 3);
        issueV(myv1, (m0 + 64) & (HW - 1));         // overlap: fetch next tile
        doChunk(m0, myv0);
        __builtin_prefetch(k + hoff + (size_t)(((m0 + 128) & (HW - 1)) + lane) * HD, 0, 3);
        issueV(myv0, (m0 + 128) & (HW - 1));
        doChunk(m0 + 64, myv1);
    }

    // output: flat layout h*131072 + n*32 + d  (== reference reshape to NCHW)
    #pragma unroll
    for (int r = 0; r < 8; ++r) {
        int n = n0 + r + 8 * (lane >> 4);
        out[hoff + (size_t)n * HD + lrow]      = acc0[r];
        out[hoff + (size_t)n * HD + 16 + lrow] = acc1[r];
    }
}

// ---------------------------------------------------------------------------
extern "C" void kernel_launch(void* const* d_in, const int* in_sizes, int n_in,
                              void* d_out, int out_size, void* d_ws, size_t ws_size,
                              hipStream_t stream) {
    const float* x     = (const float*)d_in[0];
    const float* qdw_w = (const float*)d_in[1];
    const float* qdw_b = (const float*)d_in[2];
    const float* qpw_w = (const float*)d_in[3];
    const float* qpw_b = (const float*)d_in[4];
    const float* kdw_w = (const float*)d_in[5];
    const float* kdw_b = (const float*)d_in[6];
    const float* kpw_w = (const float*)d_in[7];
    const float* kpw_b = (const float*)d_in[8];
    const float* vdw_w = (const float*)d_in[9];
    const float* vdw_b = (const float*)d_in[10];
    const float* vpw_w = (const float*)d_in[11];
    const float* vpw_b = (const float*)d_in[12];
    const float* posH  = (const float*)d_in[13];
    const float* posW  = (const float*)d_in[14];
    const int*   pidx  = (const int*)d_in[15];

    // workspace: q/k/v in f16, 1M halves (2 MB) each = 6 MB total
    f16* qf = (f16*)d_ws;
    f16* kf = qf + (1 << 20);
    f16* vf = kf + (1 << 20);

    dim3 g1(HW / 64, CCH / 64, 1);
    qkv_gemm<0><<<g1, 128, 0, stream>>>(x, qdw_w, qdw_b, qpw_w, qpw_b,
                                        posH, posW, pidx, qf);
    qkv_gemm<1><<<g1, 128, 0, stream>>>(x, kdw_w, kdw_b, kpw_w, kpw_b,
                                        posH, posW, pidx, kf);
    qkv_gemm<0><<<g1, 128, 0, stream>>>(x, vdw_w, vdw_b, vpw_w, vpw_b,
                                        posH, posW, pidx, vf);

    fused_attn<<<HW / 16, 256, 0, stream>>>(qf, kf, vf, (float*)d_out);
}
